// TopoLossMSE2D_TotalCell_62088047231146
// MI455X (gfx1250) — compile-verified, matrix operand order checked by reading, not run on hardware
//
#include <hip/hip_runtime.h>

// ---------------------------------------------------------------------------
// TopoLossMSE2D on MI455X (gfx1250, wave32).
//   Stage 1: binarize + channel-max -> chamfer init (0 / 1e10), 32 maps.
//   Stage 2: 8 sequential chamfer sweeps per map (2 iters x down/up/left/right),
//            one 512-thread block per map, LDS double-buffered carry row,
//            global_prefetch_b8 ahead of the dependent chain.
//   Stage 3: per-map min/max (uint-bit atomics; distances are non-negative).
//   Stage 4: per-window (64x64) async global->LDS window fill (b128,
//            ASYNCcnt-tracked) + bitonic sort of both EDT maps in LDS +
//            squared-diff reduction on the matrix pipe via
//            v_wmma_f32_16x16x32_f16 (B = ones => row sums in f32 accum).
//   Stage 5: deterministic final reduce -> (mean*0.005, mean).
// ---------------------------------------------------------------------------

typedef __attribute__((ext_vector_type(16))) _Float16 v16h;
typedef __attribute__((ext_vector_type(8)))  float    v8f;

#define HW   (512 * 512)
#define NMAP 32
#define SQ2F 1.41421356237309515f
#define BIGF 1e20f
#define VBIG 1e10f
#define EPSF 1e-6f

// CDNA5 async global->LDS path (guarded: also keeps the host pass clean,
// where these device builtins do not exist).
#if defined(__AMDGCN__) &&                                        \
    __has_builtin(__builtin_amdgcn_global_load_async_to_lds_b128) && \
    __has_builtin(__builtin_amdgcn_s_wait_asynccnt)
#define TL_ASYNC_LDS 1
typedef int tl_i4 __attribute__((vector_size(4 * sizeof(int))));
typedef __attribute__((address_space(1))) tl_i4* tl_gptr;  // global int4*
typedef __attribute__((address_space(3))) tl_i4* tl_lptr;  // LDS int4*
#else
#define TL_ASYNC_LDS 0
#endif

// ---------------- Stage 0: init min/max slots ------------------------------
__global__ void tl_init_kernel(unsigned* __restrict__ mm) {
    int t = threadIdx.x;
    if (t < NMAP) {
        mm[2 * t + 0] = 0x7F7FFFFFu;  // +FLT_MAX bits (min slot)
        mm[2 * t + 1] = 0x00000000u;  // 0.0f bits     (max slot)
    }
}

// ---------------- Stage 1: binarize + channel max + chamfer init -----------
__global__ __launch_bounds__(256)
void tl_binit_kernel(const float* __restrict__ pred,
                     const float* __restrict__ target,
                     float* __restrict__ dist) {
    size_t gid = (size_t)blockIdx.x * 256 + threadIdx.x;  // 0 .. 32*HW-1
    int m = (int)(gid >> 18);        // map 0..31
    int p = (int)(gid & (HW - 1));   // pixel
    float comb;
    if (m < 16) {
        size_t base = ((size_t)m * 3) << 18;
        float a = (pred[base + p] + 1.0f) * 0.5f;
        float b = (pred[base + HW + p] + 1.0f) * 0.5f;
        float c = (pred[base + 2 * HW + p] + 1.0f) * 0.5f;
        float mx = fmaxf(a, fmaxf(b, c));
        comb = (mx >= 0.7f) ? 1.0f : 0.0f;
    } else {
        size_t base = ((size_t)(m - 16) * 3) << 18;
        float a = target[base + p];
        float b = target[base + HW + p];
        float c = target[base + 2 * HW + p];
        comb = fmaxf(a, fmaxf(b, c));  // already binary 0/1
    }
    dist[gid] = VBIG * (1.0f - comb);
}

// ---------------- Stage 2: chamfer sweeps ----------------------------------
__device__ __forceinline__ float tl_relax(float cur, float pl, float pc, float pr) {
    return fminf(cur, fminf(pc + 1.0f, fminf(pl, pr) + SQ2F));
}

__global__ __launch_bounds__(512)
void tl_sweep_kernel(float* __restrict__ dist) {
    float* d = dist + (size_t)blockIdx.x * HW;
    const int t = threadIdx.x;  // 0..511: column (row sweeps) or row (col sweeps)
    __shared__ float buf[2][512];

    for (int iter = 0; iter < 2; ++iter) {
        // ---- rows, downward ----
        {
            int rd = 0;
            buf[0][t] = d[t];
            __syncthreads();
            for (int r = 1; r < 512; ++r) {
                if (r + 8 < 512) __builtin_prefetch(&d[(r + 8) * 512 + t], 0, 1);
                float pl = (t > 0)   ? buf[rd][t - 1] : BIGF;
                float pc =             buf[rd][t];
                float pr = (t < 511) ? buf[rd][t + 1] : BIGF;
                float v = tl_relax(d[r * 512 + t], pl, pc, pr);
                d[r * 512 + t] = v;
                buf[rd ^ 1][t] = v;
                rd ^= 1;
                __syncthreads();
            }
        }
        // ---- rows, upward ----
        {
            int rd = 0;
            buf[0][t] = d[511 * 512 + t];
            __syncthreads();
            for (int r = 510; r >= 0; --r) {
                if (r - 8 >= 0) __builtin_prefetch(&d[(r - 8) * 512 + t], 0, 1);
                float pl = (t > 0)   ? buf[rd][t - 1] : BIGF;
                float pc =             buf[rd][t];
                float pr = (t < 511) ? buf[rd][t + 1] : BIGF;
                float v = tl_relax(d[r * 512 + t], pl, pc, pr);
                d[r * 512 + t] = v;
                buf[rd ^ 1][t] = v;
                rd ^= 1;
                __syncthreads();
            }
        }
        // ---- columns, left -> right (t = row index) ----
        {
            int rd = 0;
            buf[0][t] = d[t * 512];
            __syncthreads();
            for (int cc = 1; cc < 512; ++cc) {
                float pl = (t > 0)   ? buf[rd][t - 1] : BIGF;
                float pc =             buf[rd][t];
                float pr = (t < 511) ? buf[rd][t + 1] : BIGF;
                float v = tl_relax(d[t * 512 + cc], pl, pc, pr);
                d[t * 512 + cc] = v;
                buf[rd ^ 1][t] = v;
                rd ^= 1;
                __syncthreads();
            }
        }
        // ---- columns, right -> left ----
        {
            int rd = 0;
            buf[0][t] = d[t * 512 + 511];
            __syncthreads();
            for (int cc = 510; cc >= 0; --cc) {
                float pl = (t > 0)   ? buf[rd][t - 1] : BIGF;
                float pc =             buf[rd][t];
                float pr = (t < 511) ? buf[rd][t + 1] : BIGF;
                float v = tl_relax(d[t * 512 + cc], pl, pc, pr);
                d[t * 512 + cc] = v;
                buf[rd ^ 1][t] = v;
                rd ^= 1;
                __syncthreads();
            }
        }
    }
}

// ---------------- Stage 3: per-map min/max ---------------------------------
__global__ __launch_bounds__(256)
void tl_minmax_kernel(const float* __restrict__ dist, unsigned* __restrict__ mm) {
    const int m   = blockIdx.x >> 6;  // map 0..31
    const int blk = blockIdx.x & 63;  // chunk 0..63 (4096 px each)
    const float* d = dist + (size_t)m * HW + (size_t)blk * 4096;
    const int t = threadIdx.x;
    float lo = 3.4e38f, hi = 0.0f;
    for (int i = t; i < 4096; i += 256) {
        float v = d[i];
        lo = fminf(lo, v);
        hi = fmaxf(hi, v);
    }
    __shared__ float slo[256], shi[256];
    slo[t] = lo; shi[t] = hi;
    __syncthreads();
    for (int s = 128; s > 0; s >>= 1) {
        if (t < s) {
            slo[t] = fminf(slo[t], slo[t + s]);
            shi[t] = fmaxf(shi[t], shi[t + s]);
        }
        __syncthreads();
    }
    if (t == 0) {
        // distances are >= 0 so uint bit order == float order
        atomicMin(&mm[2 * m + 0], __float_as_uint(slo[0]));
        atomicMax(&mm[2 * m + 1], __float_as_uint(shi[0]));
    }
}

// ---------------- Stage 4: window sort + WMMA squared-diff reduce ----------
__global__ __launch_bounds__(512)
void tl_sortmse_kernel(const float* __restrict__ dist,
                       const unsigned* __restrict__ mm,
                       float* __restrict__ wsums) {
    const int item = blockIdx.x >> 6;  // 0..15
    const int win  = blockIdx.x & 63;  // 0..63
    const int wy = win >> 3, wx = win & 7;
    const int t = threadIdx.x;

    const float* dp = dist + (size_t)item * HW;
    const float* dt = dist + (size_t)(16 + item) * HW;

    const float pmin = __uint_as_float(mm[2 * item + 0]);
    const float pmax = __uint_as_float(mm[2 * item + 1]);
    const float tmin = __uint_as_float(mm[2 * (16 + item) + 0]);
    const float tmax = __uint_as_float(mm[2 * (16 + item) + 1]);
    const float psc = 1.0f / (pmax - pmin + EPSF);
    const float tsc = 1.0f / (tmax - tmin + EPSF);

    __shared__ float sp[4096];
    __shared__ float st[4096];
    __shared__ float wsum[16];

#if TL_ASYNC_LDS
    // Direct global->LDS fill, b128 (16B-aligned: rows are 2KB apart, window
    // columns start at 256B multiples). Tracked by ASYNCcnt, no VGPR staging.
    for (int q = t; q < 1024; q += 512) {
        int i = q * 4;                 // element index, multiple of 4
        int r = i >> 6, cc = i & 63;
        int g = (wy * 64 + r) * 512 + wx * 64 + cc;
        __builtin_amdgcn_global_load_async_to_lds_b128(
            (tl_gptr)(dp + g), (tl_lptr)(sp + i), 0, 0);
        __builtin_amdgcn_global_load_async_to_lds_b128(
            (tl_gptr)(dt + g), (tl_lptr)(st + i), 0, 0);
    }
    __builtin_amdgcn_s_wait_asynccnt(0);
    __syncthreads();
    // normalize in place
    for (int i = t; i < 4096; i += 512) {
        sp[i] = (sp[i] - pmin) * psc;
        st[i] = (st[i] - tmin) * tsc;
    }
    __syncthreads();
#else
    // Fallback: load + normalize through VGPRs
    for (int i = t; i < 4096; i += 512) {
        int r = i >> 6, cc = i & 63;
        int g = (wy * 64 + r) * 512 + wx * 64 + cc;
        sp[i] = (dp[g] - pmin) * psc;
        st[i] = (dt[g] - tmin) * tsc;
    }
    __syncthreads();
#endif

    // bitonic sort both arrays (ascending), N = 4096
    for (int k = 2; k <= 4096; k <<= 1) {
        for (int j = k >> 1; j > 0; j >>= 1) {
            for (int i = t; i < 4096; i += 512) {
                int ixj = i ^ j;
                if (ixj > i) {
                    bool up = ((i & k) == 0);
                    float a = sp[i], b = sp[ixj];
                    if ((a > b) == up) { sp[i] = b; sp[ixj] = a; }
                    float c = st[i], e = st[ixj];
                    if ((c > e) == up) { st[i] = e; st[ixj] = c; }
                }
            }
            __syncthreads();
        }
    }

    // squared-diff reduction on the matrix pipe:
    // wave w (w<8) owns elements [w*512, w*512+512); A(16x32)=d^2 in f16,
    // B = ones  =>  D[i][j] = sum_k A[i][k];  sum(all D entries) = 16 * sum(A).
    const int wave = t >> 5, lane = t & 31;
    if (wave < 8) {
        v16h bones;
        #pragma unroll
        for (int k = 0; k < 16; ++k) bones[k] = (_Float16)1.0f;
        v16h a;
        const int base = wave * 512 + lane * 16;
        #pragma unroll
        for (int k = 0; k < 16; ++k) {
            float dq = sp[base + k] - st[base + k];
            a[k] = (_Float16)(dq * dq);
        }
        v8f acc = {};
        acc = __builtin_amdgcn_wmma_f32_16x16x32_f16(
            /*neg_a=*/false, a, /*neg_b=*/false, bones,
            /*c_mod=*/(short)0, acc, /*reuse_a=*/false, /*reuse_b=*/false);
        float s = acc[0] + acc[1] + acc[2] + acc[3] +
                  acc[4] + acc[5] + acc[6] + acc[7];
        #pragma unroll
        for (int off = 16; off > 0; off >>= 1) s += __shfl_down(s, off);
        if (lane == 0) wsum[wave] = s * (1.0f / 16.0f);
    }
    __syncthreads();
    if (t == 0) {
        float tot = 0.0f;
        #pragma unroll
        for (int w = 0; w < 8; ++w) tot += wsum[w];
        wsums[blockIdx.x] = tot;  // deterministic per-window partial
    }
}

// ---------------- Stage 5: deterministic final reduce ----------------------
__global__ __launch_bounds__(256)
void tl_finalize_kernel(const float* __restrict__ wsums, float* __restrict__ out) {
    __shared__ float s[256];
    int t = threadIdx.x;
    s[t] = wsums[t] + wsums[t + 256] + wsums[t + 512] + wsums[t + 768];
    __syncthreads();
    for (int k = 128; k > 0; k >>= 1) {
        if (t < k) s[t] += s[t + k];
        __syncthreads();
    }
    if (t == 0) {
        float mean = s[0] / 4194304.0f;  // 16 items * 64 windows * 4096 elems
        out[0] = mean * 0.005f;          // TOPO_WEIGHT * mean
        out[1] = mean;
    }
}

// ---------------------------------------------------------------------------
extern "C" void kernel_launch(void* const* d_in, const int* in_sizes, int n_in,
                              void* d_out, int out_size, void* d_ws, size_t ws_size,
                              hipStream_t stream) {
    (void)in_sizes; (void)n_in; (void)out_size; (void)ws_size;

    const float* pred   = (const float*)d_in[0];  // (16,3,512,512) f32
    const float* target = (const float*)d_in[1];  // (16,3,512,512) f32
    float* out = (float*)d_out;                   // 2 floats

    const size_t DIST_BYTES = (size_t)NMAP * HW * sizeof(float);  // 33.5 MB
    float*    dist  = (float*)d_ws;
    unsigned* mm    = (unsigned*)((char*)d_ws + DIST_BYTES);          // 32 x {min,max}
    float*    wsums = (float*)((char*)d_ws + DIST_BYTES + 256);       // 1024 partials

    tl_init_kernel<<<1, 32, 0, stream>>>(mm);
    tl_binit_kernel<<<(NMAP * HW) / 256, 256, 0, stream>>>(pred, target, dist);
    tl_sweep_kernel<<<NMAP, 512, 0, stream>>>(dist);
    tl_minmax_kernel<<<NMAP * 64, 256, 0, stream>>>(dist, mm);
    tl_sortmse_kernel<<<16 * 64, 512, 0, stream>>>(dist, mm, wsums);
    tl_finalize_kernel<<<1, 256, 0, stream>>>(wsums, out);
}
